// cross_attention_29008209117453
// MI455X (gfx1250) — matrix operand order repaired and checked by reading, not compiled, not run on hardware
//
#include <hip/hip_runtime.h>
#include <stdint.h>

#define Bdim 8
#define Ldim 2048
#define Hdim 1024
#define Mtot (Bdim * Ldim)   // 16384 flattened (b,l) rows

typedef __attribute__((ext_vector_type(16))) __bf16 v16bf;
typedef __attribute__((ext_vector_type(8)))  __bf16 v8bf;
typedef __attribute__((ext_vector_type(8)))  float  v8f;

union ABf {
  v16bf v;
  v8bf  h[2];
  __bf16 e[16];
};

__device__ __forceinline__ v8f wmma_bf16(v16bf a, v16bf b, v8f c) {
  // D = A(16x32 bf16) * B(32x16 bf16) + C(16x16 f32)
  return __builtin_amdgcn_wmma_f32_16x16x32_bf16(
      /*neg_a=*/false, a, /*neg_b=*/false, b,
      /*c_mod=*/(short)0, c, /*reuse_a=*/false, /*reuse_b=*/false);
}

// ---------------------------------------------------------------------------
// Projection GEMM: out[M,N](bf16) = A[M,K](f32) @ W[K,N](f32) + bias
// M = 16384, N = K = 1024. transposed=1 writes out[N][M] (for V^T).
// Workgroup: 256 thr (8 waves). Tile M=128 x N=64, K-step 32.
// A-tile staged to LDS as raw f32 with CDNA5 async-to-LDS DMA.
// ---------------------------------------------------------------------------
__global__ __launch_bounds__(256) void proj_bf16_kernel(
    const float* __restrict__ A, const float* __restrict__ W,
    const float* __restrict__ bias, __bf16* __restrict__ out, int transposed)
{
  __shared__ float  a_s[128][36];  // f32 A tile (async DMA target), pad 4
  __shared__ __bf16 w_s[64][40];   // W tile transposed [n][k] bf16, pad 8

  const int tid  = threadIdx.x;
  const int wave = tid >> 5;
  const int lane = tid & 31;
  const int half = lane >> 4;     // 0: lanes 0-15, 1: lanes 16-31
  const int ln   = lane & 15;

  const int n_blk = blockIdx.x & 15;   // 16 N-blocks of 64
  const int m_blk = blockIdx.x >> 4;   // 128 M-blocks of 128
  const int m0  = m_blk * 128;
  const int n0g = n_blk * 64;

  v8f acc[4] = {};                 // 4 n-tiles of 16x16 per wave

  const int ar_row = tid >> 1;           // 128 rows, 2 threads/row
  const int ar_col = (tid & 1) * 16;     // 16 f32 each
  const int wr_row = tid >> 3;           // 32 rows, 8 threads/row
  const int wr_col = (tid & 7) * 8;      // 8 f32 each

  for (int k0 = 0; k0 < Hdim; k0 += 32) {
    // ---- stage A tile (128x32 f32) via async DMA to LDS ----
    {
      uint64_t g = (uint64_t)(A + (size_t)(m0 + ar_row) * Hdim + k0 + ar_col);
      unsigned loff = (unsigned)(size_t)(&a_s[ar_row][ar_col]);
      asm volatile(
        "global_load_async_to_lds_b128 %0, %1, off\n\t"
        "global_load_async_to_lds_b128 %0, %1, off offset:16\n\t"
        "global_load_async_to_lds_b128 %0, %1, off offset:32\n\t"
        "global_load_async_to_lds_b128 %0, %1, off offset:48"
        :: "v"(loff), "v"(g) : "memory");
    }
    // ---- stage W tile (32x64) transposed -> bf16 LDS ----
    {
      const float* wp = W + (size_t)(k0 + wr_row) * Hdim + n0g + wr_col;
      float4 w0 = *(const float4*)wp;
      float4 w1 = *(const float4*)(wp + 4);
      w_s[wr_col + 0][wr_row] = (__bf16)w0.x;
      w_s[wr_col + 1][wr_row] = (__bf16)w0.y;
      w_s[wr_col + 2][wr_row] = (__bf16)w0.z;
      w_s[wr_col + 3][wr_row] = (__bf16)w0.w;
      w_s[wr_col + 4][wr_row] = (__bf16)w1.x;
      w_s[wr_col + 5][wr_row] = (__bf16)w1.y;
      w_s[wr_col + 6][wr_row] = (__bf16)w1.z;
      w_s[wr_col + 7][wr_row] = (__bf16)w1.w;
    }
    if (k0 + 32 < Hdim) {  // prefetch next K-tile (global_prefetch_b8)
      __builtin_prefetch(A + (size_t)(m0 + ar_row) * Hdim + (k0 + 32) + ar_col, 0, 1);
      __builtin_prefetch(W + (size_t)(k0 + 32 + wr_row) * Hdim + n0g + wr_col, 0, 1);
    }
    asm volatile("s_wait_asynccnt 0x0" ::: "memory");
    __syncthreads();

    // ---- A fragment: 16x32 bf16, contiguous K layout (half: K 0..15 / 16..31) ----
    ABf af;
    {
      const float* ap0 = &a_s[wave * 16 + ln][half * 16];
      #pragma unroll
      for (int i = 0; i < 8; ++i) af.e[i] = (__bf16)ap0[i];
      #pragma unroll
      for (int i = 0; i < 8; ++i) af.e[8 + i] = (__bf16)ap0[8 + i];
    }
    // ---- 4 B fragments + WMMA ----
    #pragma unroll
    for (int t = 0; t < 4; ++t) {
      ABf bf;
      const __bf16* bp = &w_s[t * 16 + ln][half * 16];
      bf.h[0] = *(const v8bf*)bp;
      bf.h[1] = *(const v8bf*)(bp + 8);
      acc[t] = wmma_bf16(af.v, bf.v, acc[t]);
    }
    __syncthreads();
  }

  // ---- epilogue: bias add, bf16 store (normal or transposed) ----
  #pragma unroll
  for (int t = 0; t < 4; ++t) {
    const int n = n0g + t * 16 + ln;
    const float bvv = bias[n];
    const int mrow = m0 + wave * 16 + half * 8;   // C layout: VGPR j -> row j (+8 hi half)
    if (!transposed) {
      __bf16* op = out + (size_t)mrow * Hdim + n;
      #pragma unroll
      for (int j = 0; j < 8; ++j) op[(size_t)j * Hdim] = (__bf16)(acc[t][j] + bvv);
    } else {
      ABf tmp;
      #pragma unroll
      for (int j = 0; j < 8; ++j) tmp.e[j] = (__bf16)(acc[t][j] + bvv);
      *(v8bf*)(out + (size_t)n * Mtot + mrow) = tmp.h[0];  // contiguous 8 along M
    }
  }
}

// ---------------------------------------------------------------------------
// Flash attention: per block (64 thr, 2 waves) 32 q-rows of one batch.
// O accumulates in f32 LDS; online softmax; residual + 1/l in epilogue.
// ---------------------------------------------------------------------------
__global__ __launch_bounds__(64) void attn_kernel(
    const __bf16* __restrict__ qg, const __bf16* __restrict__ kg,
    const __bf16* __restrict__ vTg, const float* __restrict__ xs,
    float* __restrict__ out)
{
  __shared__ float  O_s[32][1028];      // 131584 B f32 accumulator
  __shared__ __bf16 P_s[2][16][32];     // per-wave P relayout scratch
  __shared__ float  l_s[32];

  const int tid  = threadIdx.x;
  const int wave = tid >> 5;
  const int lane = tid & 31;
  const int half = lane >> 4;
  const int ln   = lane & 15;

  const int b  = blockIdx.x >> 6;       // 64 q-tiles per batch
  const int qt = blockIdx.x & 63;
  const int q0 = qt * 32;
  const int wr0 = wave * 16;

  for (int idx = tid * 4; idx < 32 * 1024; idx += 64 * 4) {
    const int r = idx >> 10, c = idx & 1023;
    *(float4*)&O_s[r][c] = make_float4(0.f, 0.f, 0.f, 0.f);
  }
  __syncthreads();

  float m_i[8], l_i[8], corr[8];
  #pragma unroll
  for (int j = 0; j < 8; ++j) { m_i[j] = -1e30f; l_i[j] = 0.f; }

  const float scale = 0.022097086912079608f;   // 2048^-0.5 (kv-seq-len scaling)

  const __bf16* qbase = qg + (size_t)b * Ldim * Hdim + (size_t)(q0 + wr0 + ln) * Hdim;
  const __bf16* kbase = kg + (size_t)b * Ldim * Hdim;

  for (int kv0 = 0; kv0 < Ldim; kv0 += 32) {
    // ---- S = q @ k^T for two 16-col kv tiles, K-loop over H ----
    v8f s0 = {}, s1 = {};
    const __bf16* qp  = qbase + half * 16;
    const __bf16* k0p = kbase + (size_t)(kv0 + ln) * Hdim + half * 16;
    const __bf16* k1p = kbase + (size_t)(kv0 + 16 + ln) * Hdim + half * 16;
    #pragma unroll 4
    for (int h0 = 0; h0 < Hdim; h0 += 32) {
      ABf qa, kf0, kf1;
      qa.h[0]  = *(const v8bf*)(qp + h0);
      qa.h[1]  = *(const v8bf*)(qp + h0 + 8);
      kf0.h[0] = *(const v8bf*)(k0p + h0);
      kf0.h[1] = *(const v8bf*)(k0p + h0 + 8);
      kf1.h[0] = *(const v8bf*)(k1p + h0);
      kf1.h[1] = *(const v8bf*)(k1p + h0 + 8);
      s0 = wmma_bf16(qa.v, kf0.v, s0);
      s1 = wmma_bf16(qa.v, kf1.v, s1);
    }

    // ---- online softmax: rows j (+8 for hi half) reduce across 16-lane halves ----
    #pragma unroll
    for (int j = 0; j < 8; ++j) {
      const float a0 = s0[j] * scale;
      const float a1 = s1[j] * scale;
      float bm = fmaxf(a0, a1);
      #pragma unroll
      for (int off = 1; off < 16; off <<= 1)
        bm = fmaxf(bm, __shfl_xor(bm, off, 32));
      const float mn = fmaxf(m_i[j], bm);
      const float cr = __expf(m_i[j] - mn);
      const float p0 = __expf(a0 - mn);
      const float p1 = __expf(a1 - mn);
      float ps = p0 + p1;
      #pragma unroll
      for (int off = 1; off < 16; off <<= 1)
        ps += __shfl_xor(ps, off, 32);
      l_i[j] = l_i[j] * cr + ps;
      m_i[j] = mn;
      corr[j] = cr;
      P_s[wave][half * 8 + j][ln]      = (__bf16)p0;   // C-layout -> LDS
      P_s[wave][half * 8 + j][16 + ln] = (__bf16)p1;
    }
    asm volatile("s_wait_dscnt 0x0" ::: "memory");

    // ---- reload P in A-fragment layout (16x32 bf16) ----
    ABf pf;
    pf.h[0] = *(const v8bf*)&P_s[wave][ln][half * 16];
    pf.h[1] = *(const v8bf*)&P_s[wave][ln][half * 16 + 8];

    // ---- O[16x1024] += corr * O + P @ V, tiled by 16 columns ----
    for (int n0 = 0; n0 < Hdim; n0 += 16) {
      ABf vf;
      const __bf16* vp = vTg + (size_t)(n0 + ln) * Mtot + (size_t)b * Ldim + kv0 + half * 16;
      vf.h[0] = *(const v8bf*)vp;
      vf.h[1] = *(const v8bf*)(vp + 8);
      v8f cin;
      #pragma unroll
      for (int j = 0; j < 8; ++j)
        cin[j] = corr[j] * O_s[wr0 + half * 8 + j][n0 + ln];
      v8f o = wmma_bf16(pf.v, vf.v, cin);
      #pragma unroll
      for (int j = 0; j < 8; ++j)
        O_s[wr0 + half * 8 + j][n0 + ln] = o[j];
    }
  }

  if (ln == 0) {
    #pragma unroll
    for (int j = 0; j < 8; ++j) l_s[wr0 + half * 8 + j] = l_i[j];
  }
  __syncthreads();

  // ---- epilogue: out = O / l + xs ----
  const float* xsb = xs + ((size_t)b * Ldim + q0) * Hdim;
  float* ob = out + ((size_t)b * Ldim + q0) * Hdim;
  for (int idx = tid * 4; idx < 32 * 1024; idx += 64 * 4) {
    const int r = idx >> 10, c = idx & 1023;
    const float inv = 1.0f / l_s[r];
    float4 ov = *(float4*)&O_s[r][c];
    const float4 xv = *(const float4*)(xsb + (size_t)r * Hdim + c);
    float4 res;
    res.x = ov.x * inv + xv.x;
    res.y = ov.y * inv + xv.y;
    res.z = ov.z * inv + xv.z;
    res.w = ov.w * inv + xv.w;
    *(float4*)(ob + (size_t)r * Hdim + c) = res;
  }
}

// ---------------------------------------------------------------------------
extern "C" void kernel_launch(void* const* d_in, const int* in_sizes, int n_in,
                              void* d_out, int out_size, void* d_ws, size_t ws_size,
                              hipStream_t stream)
{
  (void)in_sizes; (void)n_in; (void)out_size; (void)ws_size;
  const float* xs = (const float*)d_in[0];
  const float* x  = (const float*)d_in[1];
  const float* Wq = (const float*)d_in[2];
  const float* bq = (const float*)d_in[3];
  const float* Wk = (const float*)d_in[4];
  const float* bk = (const float*)d_in[5];
  const float* Wv = (const float*)d_in[6];
  const float* bv = (const float*)d_in[7];
  float* out = (float*)d_out;

  // workspace: q (32MB) | k (32MB) | v^T (32MB), all bf16
  __bf16* qb = (__bf16*)d_ws;
  __bf16* kb = qb + (size_t)Mtot * Hdim;
  __bf16* vT = kb + (size_t)Mtot * Hdim;

  proj_bf16_kernel<<<dim3(2048), dim3(256), 0, stream>>>(x,  Wq, bq, qb, 0);
  proj_bf16_kernel<<<dim3(2048), dim3(256), 0, stream>>>(xs, Wk, bk, kb, 0);
  proj_bf16_kernel<<<dim3(2048), dim3(256), 0, stream>>>(xs, Wv, bv, vT, 1);
  attn_kernel<<<dim3(Bdim * (Ldim / 32)), dim3(64), 0, stream>>>(qb, kb, vT, xs, out);
}
